// Decoder_57397942943858
// MI455X (gfx1250) — compile-verified, matrix operand order checked by reading, not run on hardware
//
#include <hip/hip_runtime.h>
#include <cstdint>
#include <cstddef>

// ---------------------------------------------------------------------------
// Types & WMMA helpers (CDNA5 / gfx1250, wave32)
// ---------------------------------------------------------------------------
typedef unsigned short u16;
typedef __attribute__((ext_vector_type(16))) __bf16 v16bf;
typedef __attribute__((ext_vector_type(8)))  float  v8f;

union Frag32B { v16bf v; uint4 q[2]; };

__device__ __forceinline__ u16 f2bf(float f) {
  unsigned u = __float_as_uint(f);
  u = u + 0x7FFFu + ((u >> 16) & 1u);   // round-to-nearest-even
  return (u16)(u >> 16);
}
__device__ __forceinline__ float sigmf(float x) { return 1.0f / (1.0f + __expf(-x)); }

// A fragment (16x32 bf16): lane = hf*16 + m ; elements j -> k = (j&7)+8*hf+16*(j>>3)
__device__ __forceinline__ v16bf load_afrag(const u16* row, int kb, int hf) {
  Frag32B f;
  f.q[0] = *(const uint4*)(row + kb + 8 * hf);
  f.q[1] = *(const uint4*)(row + kb + 16 + 8 * hf);
  return f.v;
}
// B fragment from pre-swizzled blocks: 32 lanes x 16 halves contiguous (512 u16 / block)
__device__ __forceinline__ v16bf load_bfrag(const u16* blk) {
  Frag32B f;
  f.q[0] = *(const uint4*)(blk);
  f.q[1] = *(const uint4*)(blk + 8);
  return f.v;
}
#define WMMA_BF16(a, b, c) \
  __builtin_amdgcn_wmma_f32_16x16x32_bf16(false, (a), false, (b), (short)0, (c), false, false)

// Model constants
#define NMEL 80
#define ENC_D 512
#define DEC_D 1024
#define PRE_D 256
#define ATTN_D 128
#define LOC_F 32
#define LOC_K 31
#define BB 32
#define SS 256
#define TT 512
#define LDA0 1792       // [prenet 256 | ctx 512 | h0 1024]
#define LDA1 2560       // [h0 1024 | ctx 512 | h1 1024]
#define KT0 56
#define KT1 80
#define G4 4096
#define NWG 64

// ---------------------------------------------------------------------------
// Weight packers: produce fragment-native swizzled bf16 blocks
//   dst[((nt*KTtot + kt)*32 + lane)*16 + j],  k = kt*32 + (j&7)+8*(lane>>4)+16*(j>>3)
// ---------------------------------------------------------------------------
__global__ void pack_linear_kernel(const float* __restrict__ W, int Ksrc, int Nsrc,
                                   int src_kn, int ktoff, int KTtot, int NT,
                                   u16* __restrict__ dst) {
  int KTl = (Ksrc + 31) >> 5;
  long total = (long)KTl * NT * 512;
  for (long g = (long)blockIdx.x * blockDim.x + threadIdx.x; g < total;
       g += (long)gridDim.x * blockDim.x) {
    int j = (int)(g & 15);
    int lane = (int)((g >> 4) & 31);
    long blk = g >> 9;
    int ktl = (int)(blk % KTl);
    int nt = (int)(blk / KTl);
    int hf = lane >> 4;
    int kloc = (j & 7) + 8 * hf + 16 * (j >> 3);
    int k = ktl * 32 + kloc;
    int n = nt * 16 + (lane & 15);
    float v = 0.f;
    if (k < Ksrc && n < Nsrc)
      v = src_kn ? W[(size_t)k * Nsrc + n] : W[(size_t)n * Ksrc + k];
    dst[(((size_t)nt * KTtot + (ktoff + ktl)) * 32 + lane) * 16 + j] = f2bf(v);
  }
}

// postnet conv weights: B[kk][n], kk = tap*Cp + ci, src w[cout][cin][5]
__global__ void pack_postnet_kernel(const float* __restrict__ w, int cout, int cin,
                                    int Cp, u16* __restrict__ dst) {
  int KT = (5 * Cp) >> 5;
  int NT = (cout + 15) >> 4;
  long total = (long)KT * NT * 512;
  for (long g = (long)blockIdx.x * blockDim.x + threadIdx.x; g < total;
       g += (long)gridDim.x * blockDim.x) {
    int j = (int)(g & 15);
    int lane = (int)((g >> 4) & 31);
    long blk = g >> 9;
    int kt = (int)(blk % KT);
    int nt = (int)(blk / KT);
    int hf = lane >> 4;
    int kloc = (j & 7) + 8 * hf + 16 * (j >> 3);
    int kk = kt * 32 + kloc;
    int n = nt * 16 + (lane & 15);
    int tap = kk / Cp, ci = kk % Cp;
    float v = 0.f;
    if (n < cout && ci < cin) v = w[((size_t)n * cin + ci) * 5 + tap];
    dst[(((size_t)nt * KT + kt) * 32 + lane) * 16 + j] = f2bf(v);
  }
}

__global__ void add_bias_kernel(const float* a, const float* b, float* d, int n) {
  int i = blockIdx.x * blockDim.x + threadIdx.x;
  if (i < n) d[i] = a[i] + b[i];
}

__global__ void f2b_kernel(const float* __restrict__ s, u16* __restrict__ d, long n) {
  for (long i = (long)blockIdx.x * blockDim.x + threadIdx.x; i < n;
       i += (long)gridDim.x * blockDim.x)
    d[i] = f2bf(s[i]);
}

// prenet input: Xp[b*512+t][96], col<80 -> (t==0 ? 0 : mels[b][t-1][col])
__global__ void build_xp_kernel(const float* __restrict__ mels, u16* __restrict__ Xp) {
  long total = (long)BB * TT * 96;
  for (long g = (long)blockIdx.x * blockDim.x + threadIdx.x; g < total;
       g += (long)gridDim.x * blockDim.x) {
    long r = g / 96;
    int c = (int)(g % 96);
    int b = (int)(r >> 9), t = (int)(r & 511);
    float v = 0.f;
    if (c < NMEL && t > 0) v = mels[((size_t)b * TT + (t - 1)) * NMEL + c];
    Xp[g] = f2bf(v);
  }
}

// postnet input: xpad0[b][516][96] bf16, halo rows zero
__global__ void build_xpad0_kernel(const float* __restrict__ melsout, u16* __restrict__ x0) {
  long total = (long)BB * 516 * 96;
  for (long g = (long)blockIdx.x * blockDim.x + threadIdx.x; g < total;
       g += (long)gridDim.x * blockDim.x) {
    long rem = g;
    int c = (int)(rem % 96); rem /= 96;
    int r = (int)(rem % 516);
    int b = (int)(rem / 516);
    int t = r - 2;
    float v = 0.f;
    if (c < NMEL && t >= 0 && t < TT) v = melsout[((size_t)b * TT + t) * NMEL + c];
    x0[g] = f2bf(v);
  }
}

__global__ void zero_halo_kernel(u16* xa, u16* xb) {
  long total = 2L * BB * 4 * 512;
  for (long g = (long)blockIdx.x * blockDim.x + threadIdx.x; g < total;
       g += (long)gridDim.x * blockDim.x) {
    int which = (int)(g >> 16);
    int rem = (int)(g & 65535);
    int b = rem >> 11;
    int hr = (rem >> 9) & 3;
    int c = rem & 511;
    int r = (hr < 2) ? hr : (512 + hr);
    (which ? xb : xa)[((size_t)b * 516 + r) * 512 + c] = 0;
  }
}

// ---------------------------------------------------------------------------
// Generic WMMA GEMM: C[M][N] = A[M][lda] * Bpacked + bias ; 8 waves = 2x4 tiles/WG
//   mode 0: fp32 out   mode 1: relu -> bf16 out
// ---------------------------------------------------------------------------
__global__ void __launch_bounds__(256)
wmma_gemm_kernel(const u16* __restrict__ A, int lda, const u16* __restrict__ Bp, int KT,
                 int M, int N, const float* __restrict__ bias, int mode,
                 float* __restrict__ Cf, int ldcf, u16* __restrict__ Cb, int ldcb) {
  int lane = threadIdx.x & 31, wave = threadIdx.x >> 5;
  int mt = blockIdx.y * 2 + (wave & 1);
  int nt = blockIdx.x * 4 + (wave >> 1);
  int ntot = (N + 15) >> 4, mtot = (M + 15) >> 4;
  if (mt >= mtot || nt >= ntot) return;
  int hf = lane >> 4;
  const u16* Arow = A + (size_t)(mt * 16 + (lane & 15)) * lda;
  const u16* Bblk = Bp + (size_t)nt * KT * 512 + lane * 16;
  v8f acc = {0.f, 0.f, 0.f, 0.f, 0.f, 0.f, 0.f, 0.f};
#pragma unroll 4
  for (int kt = 0; kt < KT; ++kt) {
    __builtin_prefetch(Bblk + 512, 0, 1);
    v16bf fa = load_afrag(Arow, kt * 32, hf);
    v16bf fb = load_bfrag(Bblk);
    Bblk += 512;
    acc = WMMA_BF16(fa, fb, acc);
  }
  int n = nt * 16 + (lane & 15);
  float bb = bias ? bias[n] : 0.f;
#pragma unroll
  for (int i = 0; i < 8; ++i) {
    int m = mt * 16 + i + 8 * hf;
    float y = acc[i] + bb;
    if (mode == 0) Cf[(size_t)m * ldcf + n] = y;
    else Cb[(size_t)m * ldcb + n] = f2bf(fmaxf(y, 0.f));
  }
}

// ---------------------------------------------------------------------------
// Postnet implicit-GEMM conv1d(k=5,SAME) + bias + BN (+tanh) on [B][T+4][Cp] bf16
// ---------------------------------------------------------------------------
__global__ void __launch_bounds__(256)
postnet_gemm_kernel(const u16* __restrict__ xin, int Cp, const u16* __restrict__ Bp, int KT,
                    int N, const float* __restrict__ bias, const float* __restrict__ gamma,
                    const float* __restrict__ beta, const float* __restrict__ mean,
                    const float* __restrict__ var, int do_tanh, int final_layer,
                    u16* __restrict__ xout, float* __restrict__ mel_res,
                    const unsigned char* __restrict__ dmask) {
  int lane = threadIdx.x & 31, wave = threadIdx.x >> 5;
  int mt = blockIdx.y * 2 + (wave & 1);            // 0..1023
  int nt = blockIdx.x * 4 + (wave >> 1);
  int ntot = (N + 15) >> 4;
  if (nt >= ntot) return;
  int hf = lane >> 4;
  int row = mt * 16 + (lane & 15);                 // m = b*512 + t
  int b = row >> 9, t = row & 511;
  const u16* xrow = xin + ((size_t)b * 516 + t) * Cp;   // tap rows are +Cp apart
  const u16* Bblk = Bp + (size_t)nt * KT * 512 + lane * 16;
  v8f acc = {0.f, 0.f, 0.f, 0.f, 0.f, 0.f, 0.f, 0.f};
#pragma unroll 4
  for (int kt = 0; kt < KT; ++kt) {
    __builtin_prefetch(Bblk + 512, 0, 1);
    int kb = kt * 32;
    int tap = kb / Cp, ci0 = kb % Cp;
    const u16* p = xrow + (size_t)tap * Cp + ci0;
    Frag32B fa;
    fa.q[0] = *(const uint4*)(p + 8 * hf);
    fa.q[1] = *(const uint4*)(p + 16 + 8 * hf);
    v16bf fb = load_bfrag(Bblk);
    Bblk += 512;
    acc = WMMA_BF16(fa.v, fb, acc);
  }
  int n = nt * 16 + (lane & 15);
  float bi = bias[n];
  float sc = rsqrtf(var[n] + 1e-5f) * gamma[n];
  float be = beta[n], mu = mean[n];
#pragma unroll
  for (int i = 0; i < 8; ++i) {
    int m = mt * 16 + i + 8 * hf;
    int b2 = m >> 9, t2 = m & 511;
    float y = (acc[i] + bi - mu) * sc + be;
    if (do_tanh) y = tanhf(y);
    if (final_layer)
      mel_res[((size_t)b2 * TT + t2) * NMEL + n] = dmask[b2 * TT + t2] ? 0.f : y;
    else
      xout[((size_t)b2 * 516 + t2 + 2) * 512 + n] = f2bf(y);
  }
}

// ---------------------------------------------------------------------------
// Persistent scan kernel: 512 sequential decoder steps, device-wide barrier
// ---------------------------------------------------------------------------
__device__ __forceinline__ void grid_sync(int* cnt, int target) {
  __syncthreads();
  if (threadIdx.x == 0) {
    __threadfence();
    atomicAdd(cnt, 1);
    while (atomicAdd(cnt, 0) < target) { __builtin_amdgcn_s_sleep(2); }
    __threadfence();
  }
  __syncthreads();
}

__global__ void __launch_bounds__(256)
scan_kernel(const u16* __restrict__ pren, const float* __restrict__ enc,
            const float* __restrict__ pmem,
            const u16* __restrict__ W0T, const float* __restrict__ bias0,
            const u16* __restrict__ W1T, const float* __restrict__ bias1,
            const u16* __restrict__ MPT, const float* __restrict__ mel_b,
            const float* __restrict__ Wq, const float* __restrict__ lconv,
            const float* __restrict__ Wloc, const float* __restrict__ vvec,
            const float* __restrict__ stop_w, const float* __restrict__ stop_b,
            const unsigned char* __restrict__ emask, const unsigned char* __restrict__ dmask,
            u16* __restrict__ A0, u16* __restrict__ A1,
            float* __restrict__ gates, float* __restrict__ h0, float* __restrict__ c0,
            float* __restrict__ h1, float* __restrict__ c1, float* __restrict__ ctx,
            float* __restrict__ aw, float* __restrict__ cw, float* __restrict__ locb,
            float* __restrict__ out_mel, float* __restrict__ out_stop,
            float* __restrict__ out_attn, int* __restrict__ syncc) {
  const int tid = threadIdx.x, wg = blockIdx.x;
  const int lane = tid & 31, wave = tid >> 5;
  const int gthreads = NWG * 256;
  const int gid = wg * 256 + tid;
  const int hf = lane >> 4;
  int snc = 0;
  __shared__ float sm_pq[ATTN_D];
  __shared__ float sm_e[SS];
  __shared__ float sm_red[SS];

  // ---- init state + first prenet step
  for (int i = gid; i < BB * DEC_D; i += gthreads) { h0[i] = 0.f; c0[i] = 0.f; h1[i] = 0.f; c1[i] = 0.f; }
  for (int i = gid; i < BB * SS; i += gthreads) { aw[i] = 0.f; cw[i] = 0.f; }
  for (int i = gid; i < BB * ENC_D; i += gthreads) ctx[i] = 0.f;
  for (int i = gid; i < BB * LDA0; i += gthreads) A0[i] = 0;
  for (int i = gid; i < BB * LDA1; i += gthreads) A1[i] = 0;
  for (int i = gid; i < BB * PRE_D; i += gthreads) {
    int b = i >> 8, c = i & 255;
    A0[b * LDA0 + c] = pren[((size_t)b * TT) * PRE_D + c];
  }
  grid_sync(syncc, (++snc) * NWG);

#pragma unroll 1
  for (int t = 0; t < TT; ++t) {
    // ---- stage 2: gates = A0 @ W0T + bias0   (512 tiles == 512 waves)
    {
      int mt = wave & 1;
      int nt = wg * 4 + (wave >> 1);
      const u16* Arow = A0 + (size_t)(mt * 16 + (lane & 15)) * LDA0;
      const u16* Bblk = W0T + (size_t)nt * KT0 * 512 + lane * 16;
      v8f acc = {0.f, 0.f, 0.f, 0.f, 0.f, 0.f, 0.f, 0.f};
#pragma unroll 4
      for (int kt = 0; kt < KT0; ++kt) {
        __builtin_prefetch(Bblk + 512, 0, 1);
        v16bf fa = load_afrag(Arow, kt * 32, hf);
        v16bf fb = load_bfrag(Bblk);
        Bblk += 512;
        acc = WMMA_BF16(fa, fb, acc);
      }
      int n = nt * 16 + (lane & 15);
      float bb = bias0[n];
#pragma unroll
      for (int i = 0; i < 8; ++i) {
        int m = mt * 16 + i + 8 * hf;
        gates[(size_t)m * G4 + n] = acc[i] + bb;
      }
    }
    grid_sync(syncc, (++snc) * NWG);

    // ---- stage 3: LSTM0 elementwise + location conv
    for (int i = gid; i < BB * DEC_D; i += gthreads) {
      int b = i >> 10, u = i & 1023;
      const float* g = gates + (size_t)b * G4;
      float cc = sigmf(g[1024 + u]) * c0[i] + sigmf(g[u]) * tanhf(g[2048 + u]);
      float hh = sigmf(g[3072 + u]) * tanhf(cc);
      c0[i] = cc; h0[i] = hh;
      u16 hb = f2bf(hh);
      A0[(size_t)b * LDA0 + 768 + u] = hb;
      A1[(size_t)b * LDA1 + u] = hb;
    }
    for (int i = gid; i < BB * LOC_F * SS; i += gthreads) {
      int s = i & 255, f = (i >> 8) & 31, b = i >> 13;
      const float* k0 = lconv + f * 62;           // [f][2][31]
      float accv = 0.f;
      for (int k = 0; k < LOC_K; ++k) {
        int sp = s + k - 15;
        if (sp >= 0 && sp < SS)
          accv += aw[b * SS + sp] * k0[k] + cw[b * SS + sp] * k0[31 + k];
      }
      locb[i] = accv;                             // [b][f][s]
    }
    grid_sync(syncc, (++snc) * NWG);

    // ---- stage 4: attention (one WG per batch row)
    if (wg < BB) {
      int b = wg;
      for (int a = tid; a < ATTN_D; a += 256) {
        const float* hr = h0 + (size_t)b * DEC_D;
        float s0 = 0.f;
        for (int k = 0; k < DEC_D; ++k) s0 += hr[k] * Wq[k * ATTN_D + a];
        sm_pq[a] = s0;
      }
      __syncthreads();
      {
        int s = tid;
        float lf[LOC_F];
        for (int f = 0; f < LOC_F; ++f) lf[f] = locb[((size_t)b * LOC_F + f) * SS + s];
        const float* pm = pmem + ((size_t)b * SS + s) * ATTN_D;
        float e = 0.f;
        for (int a = 0; a < ATTN_D; ++a) {
          float pl = 0.f;
          for (int f = 0; f < LOC_F; ++f) pl += lf[f] * Wloc[f * ATTN_D + a];
          e += vvec[a] * tanhf(sm_pq[a] + pl + pm[a]);
        }
        if (emask[b * SS + s]) e += -1e9f;
        sm_e[s] = e;
      }
      __syncthreads();
      sm_red[tid] = sm_e[tid];
      __syncthreads();
      for (int off = 128; off > 0; off >>= 1) {
        if (tid < off) sm_red[tid] = fmaxf(sm_red[tid], sm_red[tid + off]);
        __syncthreads();
      }
      float mx = sm_red[0];
      __syncthreads();
      float ex = __expf(sm_e[tid] - mx);
      sm_red[tid] = ex;
      __syncthreads();
      for (int off = 128; off > 0; off >>= 1) {
        if (tid < off) sm_red[tid] += sm_red[tid + off];
        __syncthreads();
      }
      float a2 = ex / sm_red[0];
      sm_e[tid] = a2;                              // reuse as aw2
      aw[b * SS + tid] = a2;
      cw[b * SS + tid] += a2;
      out_attn[((size_t)b * TT + t) * SS + tid] = dmask[b * TT + t] ? 0.f : a2;
      __syncthreads();
      for (int e2 = tid; e2 < ENC_D; e2 += 256) {
        const float* eb = enc + (size_t)b * SS * ENC_D + e2;
        float s0 = 0.f;
        for (int s = 0; s < SS; ++s) s0 += sm_e[s] * eb[(size_t)s * ENC_D];
        ctx[b * ENC_D + e2] = s0;
        u16 cb = f2bf(s0);
        A0[(size_t)b * LDA0 + 256 + e2] = cb;
        A1[(size_t)b * LDA1 + 1024 + e2] = cb;
      }
    }
    grid_sync(syncc, (++snc) * NWG);

    // ---- stage 5: gates = A1 @ W1T + bias1
    {
      int mt = wave & 1;
      int nt = wg * 4 + (wave >> 1);
      const u16* Arow = A1 + (size_t)(mt * 16 + (lane & 15)) * LDA1;
      const u16* Bblk = W1T + (size_t)nt * KT1 * 512 + lane * 16;
      v8f acc = {0.f, 0.f, 0.f, 0.f, 0.f, 0.f, 0.f, 0.f};
#pragma unroll 4
      for (int kt = 0; kt < KT1; ++kt) {
        __builtin_prefetch(Bblk + 512, 0, 1);
        v16bf fa = load_afrag(Arow, kt * 32, hf);
        v16bf fb = load_bfrag(Bblk);
        Bblk += 512;
        acc = WMMA_BF16(fa, fb, acc);
      }
      int n = nt * 16 + (lane & 15);
      float bb = bias1[n];
#pragma unroll
      for (int i = 0; i < 8; ++i) {
        int m = mt * 16 + i + 8 * hf;
        gates[(size_t)m * G4 + n] = acc[i] + bb;
      }
    }
    grid_sync(syncc, (++snc) * NWG);

    // ---- stage 6: LSTM1 elementwise
    for (int i = gid; i < BB * DEC_D; i += gthreads) {
      int b = i >> 10, u = i & 1023;
      const float* g = gates + (size_t)b * G4;
      float cc = sigmf(g[1024 + u]) * c1[i] + sigmf(g[u]) * tanhf(g[2048 + u]);
      float hh = sigmf(g[3072 + u]) * tanhf(cc);
      c1[i] = cc; h1[i] = hh;
      A1[(size_t)b * LDA1 + 1536 + u] = f2bf(hh);
    }
    grid_sync(syncc, (++snc) * NWG);

    // ---- stage 7: mel projection (WMMA), stop token, prenet for t+1
    {
      int tile = wg * 8 + wave;
      if (tile < 10) {
        int mt = tile / 5, nt = tile % 5;
        const u16* Arow = A1 + (size_t)(mt * 16 + (lane & 15)) * LDA1;
        const u16* Bblk = MPT + (size_t)nt * 48 * 512 + lane * 16;
        v8f acc = {0.f, 0.f, 0.f, 0.f, 0.f, 0.f, 0.f, 0.f};
#pragma unroll 4
        for (int kt = 0; kt < 48; ++kt) {
          int kb = kt * 32;
          int base = (kb < 1024) ? 1536 : 0;       // feat = [h1 | ctx]
          Frag32B fa;
          fa.q[0] = *(const uint4*)(Arow + base + kb + 8 * hf);
          fa.q[1] = *(const uint4*)(Arow + base + kb + 16 + 8 * hf);
          v16bf fb = load_bfrag(Bblk);
          Bblk += 512;
          acc = WMMA_BF16(fa.v, fb, acc);
        }
        int n = nt * 16 + (lane & 15);
        float bb = mel_b[n];
#pragma unroll
        for (int i = 0; i < 8; ++i) {
          int m = mt * 16 + i + 8 * hf;
          float y = acc[i] + bb;
          out_mel[((size_t)m * TT + t) * NMEL + n] = dmask[m * TT + t] ? 0.f : y;
        }
      } else if (tile == 10) {
        int b = lane;
        const float* hr = h1 + (size_t)b * DEC_D;
        const float* cr = ctx + (size_t)b * ENC_D;
        float s0 = stop_b[0];
        for (int k = 0; k < DEC_D; ++k) s0 += hr[k] * stop_w[k];
        for (int k = 0; k < ENC_D; ++k) s0 += cr[k] * stop_w[1024 + k];
        out_stop[b * TT + t] = dmask[b * TT + t] ? 1000.f : s0;
      }
      if (t + 1 < TT) {
        for (int i = gid; i < BB * PRE_D; i += gthreads) {
          int b = i >> 8, c = i & 255;
          A0[(size_t)b * LDA0 + c] = pren[((size_t)b * TT + (t + 1)) * PRE_D + c];
        }
      }
    }
    grid_sync(syncc, (++snc) * NWG);
  }
}

// ---------------------------------------------------------------------------
// Host: input resolution (two possible flattening orders) + launch schedule
// ---------------------------------------------------------------------------
struct Ptrs {
  const float *enc, *mels;
  const unsigned char *emask, *dmask;
  const float *pw0, *pw1;
  const float *Wih0, *Whh0, *bih0, *bhh0;
  const float *Wih1, *Whh1, *bih1, *bhh1;
  const float *Wq, *Wm, *lconv, *Wloc, *vv;
  const float *mpw, *mpb, *spw, *spb;
  const float *pnw[5], *pnb[5], *png[5], *pnbe[5], *pnm[5], *pnv[5];
};

static int gblk(long n) {
  long b = (n + 255) / 256;
  if (b > 16384) b = 16384;
  if (b < 1) b = 1;
  return (int)b;
}

extern "C" void kernel_launch(void* const* d_in, const int* in_sizes, int n_in,
                              void* d_out, int out_size, void* d_ws, size_t ws_size,
                              hipStream_t stream) {
  (void)n_in; (void)out_size; (void)ws_size;
#define FIN(i) ((const float*)d_in[(i)])
#define BIN(i) ((const unsigned char*)d_in[(i)])
  Ptrs P;
  if (in_sizes[0] == 4194304) {   // dict-insertion order
    P.enc = FIN(0); P.emask = BIN(1); P.mels = FIN(2); P.dmask = BIN(3);
    P.pw0 = FIN(4); P.pw1 = FIN(5);
    P.Wih0 = FIN(6); P.Whh0 = FIN(7); P.bih0 = FIN(8); P.bhh0 = FIN(9);
    P.Wih1 = FIN(10); P.Whh1 = FIN(11); P.bih1 = FIN(12); P.bhh1 = FIN(13);
    P.Wq = FIN(14); P.Wm = FIN(15); P.lconv = FIN(16); P.Wloc = FIN(17); P.vv = FIN(18);
    P.mpw = FIN(19); P.mpb = FIN(20); P.spw = FIN(21); P.spb = FIN(22);
    for (int l = 0; l < 5; ++l) {
      int o = 23 + l * 6;
      P.pnw[l] = FIN(o); P.pnb[l] = FIN(o + 1); P.png[l] = FIN(o + 2);
      P.pnbe[l] = FIN(o + 3); P.pnm[l] = FIN(o + 4); P.pnv[l] = FIN(o + 5);
    }
  } else {                        // jax sorted-key pytree order
    P.dmask = BIN(0); P.emask = BIN(1); P.enc = FIN(2); P.mels = FIN(3);
    P.Wloc = FIN(4); P.Wm = FIN(5); P.Wq = FIN(6); P.lconv = FIN(7); P.vv = FIN(8);
    P.Whh0 = FIN(9); P.Wih0 = FIN(10); P.bhh0 = FIN(11); P.bih0 = FIN(12);
    P.Whh1 = FIN(13); P.Wih1 = FIN(14); P.bhh1 = FIN(15); P.bih1 = FIN(16);
    P.mpb = FIN(17); P.mpw = FIN(18);
    for (int l = 0; l < 5; ++l) {
      int o = 19 + l * 6;
      P.pnb[l] = FIN(o); P.pnbe[l] = FIN(o + 1); P.png[l] = FIN(o + 2);
      P.pnm[l] = FIN(o + 3); P.pnv[l] = FIN(o + 4); P.pnw[l] = FIN(o + 5);
    }
    P.pw0 = FIN(49); P.pw1 = FIN(50); P.spb = FIN(51); P.spw = FIN(52);
  }

  // workspace carve-up (256B aligned)
  char* ws = (char*)d_ws;
  size_t o = 0;
  auto nxt = [&](size_t bytes) { size_t r = o; o += (bytes + 255) & ~(size_t)255; return r; };
  size_t oW0T = nxt(56ULL * 256 * 512 * 2);
  size_t oW1T = nxt(80ULL * 256 * 512 * 2);
  size_t oWMT = nxt(16ULL * 8 * 512 * 2);
  size_t oMPT = nxt(48ULL * 5 * 512 * 2);
  size_t oPN0 = nxt(15ULL * 32 * 512 * 2);
  size_t oPN1 = nxt(80ULL * 32 * 512 * 2);
  size_t oPN2 = nxt(80ULL * 32 * 512 * 2);
  size_t oPN3 = nxt(80ULL * 32 * 512 * 2);
  size_t oPN4 = nxt(80ULL * 5 * 512 * 2);
  size_t oPW0 = nxt(3ULL * 16 * 512 * 2);
  size_t oPW1 = nxt(8ULL * 16 * 512 * 2);
  size_t oB0 = nxt(4096 * 4);
  size_t oB1 = nxt(4096 * 4);
  size_t oXP = nxt(16384ULL * 96 * 2);
  size_t oH1P = nxt(16384ULL * 256 * 2);
  size_t oPRN = nxt(16384ULL * 256 * 2);
  size_t oENCB = nxt(8192ULL * 512 * 2);
  size_t oPMEM = nxt(8192ULL * 128 * 4);
  size_t oA0 = nxt(32ULL * LDA0 * 2);
  size_t oA1 = nxt(32ULL * LDA1 * 2);
  size_t oGT = nxt(32ULL * 4096 * 4);
  size_t oH0 = nxt(32ULL * 1024 * 4);
  size_t oC0 = nxt(32ULL * 1024 * 4);
  size_t oH1 = nxt(32ULL * 1024 * 4);
  size_t oC1 = nxt(32ULL * 1024 * 4);
  size_t oCTX = nxt(32ULL * 512 * 4);
  size_t oAW = nxt(32ULL * 256 * 4);
  size_t oCW = nxt(32ULL * 256 * 4);
  size_t oLOC = nxt(32ULL * 32 * 256 * 4);
  size_t oXQ0 = nxt(32ULL * 516 * 96 * 2);
  size_t oXQA = nxt(32ULL * 516 * 512 * 2);
  size_t oXQB = nxt(32ULL * 516 * 512 * 2);
  size_t oSYN = nxt(256);
#define WF(off) ((float*)(ws + (off)))
#define WU(off) ((u16*)(ws + (off)))

  float* out = (float*)d_out;
  float* out_mel = out;                        // [32][512][80]
  float* out_res = out + 1310720;              // [32][512][80]
  float* out_stop = out + 2621440;             // [32][512]
  float* out_attn = out + 2637824;             // [32][512][256]

  hipMemsetAsync(ws + oSYN, 0, 256, stream);

  // ---- pack weights into fragment-native bf16 blocks
  pack_linear_kernel<<<gblk(24L * 256 * 512), 256, 0, stream>>>(P.Wih0, 768, 4096, 0, 0, 56, 256, WU(oW0T));
  pack_linear_kernel<<<gblk(32L * 256 * 512), 256, 0, stream>>>(P.Whh0, 1024, 4096, 0, 24, 56, 256, WU(oW0T));
  pack_linear_kernel<<<gblk(48L * 256 * 512), 256, 0, stream>>>(P.Wih1, 1536, 4096, 0, 0, 80, 256, WU(oW1T));
  pack_linear_kernel<<<gblk(32L * 256 * 512), 256, 0, stream>>>(P.Whh1, 1024, 4096, 0, 48, 80, 256, WU(oW1T));
  pack_linear_kernel<<<gblk(16L * 8 * 512), 256, 0, stream>>>(P.Wm, 512, 128, 1, 0, 16, 8, WU(oWMT));
  pack_linear_kernel<<<gblk(48L * 5 * 512), 256, 0, stream>>>(P.mpw, 1536, 80, 1, 0, 48, 5, WU(oMPT));
  pack_linear_kernel<<<gblk(3L * 16 * 512), 256, 0, stream>>>(P.pw0, 80, 256, 1, 0, 3, 16, WU(oPW0));
  pack_linear_kernel<<<gblk(8L * 16 * 512), 256, 0, stream>>>(P.pw1, 256, 256, 1, 0, 8, 16, WU(oPW1));
  pack_postnet_kernel<<<gblk(15L * 32 * 512), 256, 0, stream>>>(P.pnw[0], 512, 80, 96, WU(oPN0));
  pack_postnet_kernel<<<gblk(80L * 32 * 512), 256, 0, stream>>>(P.pnw[1], 512, 512, 512, WU(oPN1));
  pack_postnet_kernel<<<gblk(80L * 32 * 512), 256, 0, stream>>>(P.pnw[2], 512, 512, 512, WU(oPN2));
  pack_postnet_kernel<<<gblk(80L * 32 * 512), 256, 0, stream>>>(P.pnw[3], 512, 512, 512, WU(oPN3));
  pack_postnet_kernel<<<gblk(80L * 5 * 512), 256, 0, stream>>>(P.pnw[4], 80, 512, 512, WU(oPN4));
  add_bias_kernel<<<16, 256, 0, stream>>>(P.bih0, P.bhh0, WF(oB0), 4096);
  add_bias_kernel<<<16, 256, 0, stream>>>(P.bih1, P.bhh1, WF(oB1), 4096);

  // ---- parallel pre-phases
  build_xp_kernel<<<gblk(16384L * 96), 256, 0, stream>>>(P.mels, WU(oXP));
  f2b_kernel<<<gblk(8192L * 512), 256, 0, stream>>>(P.enc, WU(oENCB), 8192L * 512);
  {  // prenet layer 1: (16384x96)@(96x256) relu -> bf16
    dim3 g(4, 512);
    wmma_gemm_kernel<<<g, 256, 0, stream>>>(WU(oXP), 96, WU(oPW0), 3, 16384, 256,
                                            nullptr, 1, nullptr, 0, WU(oH1P), 256);
    // prenet layer 2
    wmma_gemm_kernel<<<g, 256, 0, stream>>>(WU(oH1P), 256, WU(oPW1), 8, 16384, 256,
                                            nullptr, 1, nullptr, 0, WU(oPRN), 256);
  }
  {  // processed_mem: (8192x512)@(512x128) -> fp32
    dim3 g(2, 256);
    wmma_gemm_kernel<<<g, 256, 0, stream>>>(WU(oENCB), 512, WU(oWMT), 16, 8192, 128,
                                            nullptr, 0, WF(oPMEM), 128, nullptr, 0);
  }

  // ---- sequential decoder scan (persistent, device-wide barrier)
  scan_kernel<<<NWG, 256, 0, stream>>>(
      WU(oPRN), P.enc, WF(oPMEM), WU(oW0T), WF(oB0), WU(oW1T), WF(oB1), WU(oMPT),
      P.mpb, P.Wq, P.lconv, P.Wloc, P.vv, P.spw, P.spb, P.emask, P.dmask,
      WU(oA0), WU(oA1), WF(oGT), WF(oH0), WF(oC0), WF(oH1), WF(oC1), WF(oCTX),
      WF(oAW), WF(oCW), WF(oLOC), out_mel, out_stop, out_attn, (int*)(ws + oSYN));

  // ---- postnet (implicit-GEMM conv stack)
  build_xpad0_kernel<<<gblk(32L * 516 * 96), 256, 0, stream>>>(out_mel, WU(oXQ0));
  zero_halo_kernel<<<gblk(2L * 32 * 4 * 512), 256, 0, stream>>>(WU(oXQA), WU(oXQB));
  {
    dim3 g8(8, 512), g2(2, 512);
    postnet_gemm_kernel<<<g8, 256, 0, stream>>>(WU(oXQ0), 96, WU(oPN0), 15, 512,
        P.pnb[0], P.png[0], P.pnbe[0], P.pnm[0], P.pnv[0], 1, 0, WU(oXQA), nullptr, P.dmask);
    postnet_gemm_kernel<<<g8, 256, 0, stream>>>(WU(oXQA), 512, WU(oPN1), 80, 512,
        P.pnb[1], P.png[1], P.pnbe[1], P.pnm[1], P.pnv[1], 1, 0, WU(oXQB), nullptr, P.dmask);
    postnet_gemm_kernel<<<g8, 256, 0, stream>>>(WU(oXQB), 512, WU(oPN2), 80, 512,
        P.pnb[2], P.png[2], P.pnbe[2], P.pnm[2], P.pnv[2], 1, 0, WU(oXQA), nullptr, P.dmask);
    postnet_gemm_kernel<<<g8, 256, 0, stream>>>(WU(oXQA), 512, WU(oPN3), 80, 512,
        P.pnb[3], P.png[3], P.pnbe[3], P.pnm[3], P.pnv[3], 1, 0, WU(oXQB), nullptr, P.dmask);
    postnet_gemm_kernel<<<g2, 256, 0, stream>>>(WU(oXQB), 512, WU(oPN4), 80, 80,
        P.pnb[4], P.png[4], P.pnbe[4], P.pnm[4], P.pnv[4], 0, 1, nullptr, out_res, P.dmask);
  }
}
